// AddAttention_53386443489409
// MI455X (gfx1250) — compile-verified, matrix operand order checked by reading, not run on hardware
//
#include <hip/hip_runtime.h>
#include <hip/hip_bf16.h>

// Problem constants from the reference
#define Bsz 4
#define LQ  512
#define LK  512
#define QS  512
#define Hd  256
#define DV  512

typedef __attribute__((ext_vector_type(16))) _Float16 v16h;
typedef __attribute__((ext_vector_type(8)))  _Float16 v8h;
typedef __attribute__((ext_vector_type(8)))  float    v8f;
typedef __attribute__((ext_vector_type(4)))  float    v4f;

// ---- LDS layout (byte offsets into dynamic shared) -------------------------
// attn   : f16 [16*512]                  @ 0       (16384 B)  lives to end
// red    : f32 [256]                     @ 16384   ( 1024 B)
// rowstat: f32 [16]                      @ 17408   (   64 B)
// overlay @ 17472:
//   phase 1/2: qr f32[16*256] | wvl f32[256] | ktile f32[32*260] | sc f32[16*512]
//   phase 3  : vtile f32[32*512]  (64 KB, reuses dead region)
#define ATTN_OFF    0
#define RED_OFF     16384
#define ROWSTAT_OFF 17408
#define OVL_OFF     17472
#define QR_OFF      (OVL_OFF)
#define WVL_OFF     (QR_OFF + 16 * 256 * 4)
#define KTILE_OFF   (WVL_OFF + 256 * 4)
#define KT_STRIDE   260                         /* 16B-aligned rows, bank-rotating */
#define SC_OFF      (KTILE_OFF + 32 * KT_STRIDE * 4)
#define VTILE_OFF   (OVL_OFF)
#define SMEM_BYTES  (SC_OFF + 16 * 512 * 4)     /* 100928 B */

__device__ __forceinline__ float fast_tanh(float x) {
#if __has_builtin(__builtin_amdgcn_tanhf)
    return __builtin_amdgcn_tanhf(x);           // gfx1250 v_tanh_f32 (TRANS pipe)
#else
    return tanhf(x);
#endif
}

// Async 16B global->LDS copy (ASYNCcnt-tracked). lds_off is a byte offset
// within the workgroup's LDS allocation (dynamic smem starts at 0 here).
__device__ __forceinline__ void async_b128(unsigned lds_off, const void* g) {
    asm volatile("global_load_async_to_lds_b128 %0, %1, off"
                 :: "v"(lds_off), "v"((unsigned long long)(uintptr_t)g)
                 : "memory");
}
__device__ __forceinline__ void wait_async0() {
#if __has_builtin(__builtin_amdgcn_s_wait_asynccnt)
    __builtin_amdgcn_s_wait_asynccnt(0);
#else
    asm volatile("s_wait_asynccnt 0x0" ::: "memory");
#endif
}

// ---------------------------------------------------------------------------
// Kernel 1: Y[m, n] = sum_d X[m, d] * W[n, d] + bias[n]
// One wave per 16x16 output tile, 8 waves per block; v_wmma_f32_16x16x32_f16.
// ---------------------------------------------------------------------------
__global__ __launch_bounds__(256) void proj_kernel(
    const float* __restrict__ X,      // [2048, 512]
    const float* __restrict__ W,      // [256, 512]
    const float* __restrict__ bias,   // [256]
    float* __restrict__ Y)            // [2048, 256]
{
    const int lane = threadIdx.x & 31;
    const int wave = threadIdx.x >> 5;
    const int n0   = blockIdx.x * 16;
    const int m0   = blockIdx.y * 128 + wave * 16;

    const int mrow = m0 + (lane & 15);
    const int ncol = n0 + (lane & 15);
    const int kb   = (lane >> 4) * 8;

    v8f c = {};
    for (int k0 = 0; k0 < QS; k0 += 32) {
        const float* ap = X + mrow * QS + k0 + kb;
        const float* bp = W + ncol * QS + k0 + kb;   // B[K][n] = W[n][K]
        v16h a, b;
#pragma unroll
        for (int i = 0; i < 8; ++i) {
            a[i]     = (_Float16)ap[i];
            a[i + 8] = (_Float16)ap[i + 16];
            b[i]     = (_Float16)bp[i];
            b[i + 8] = (_Float16)bp[i + 16];
        }
        c = __builtin_amdgcn_wmma_f32_16x16x32_f16(
                false, a, false, b, (short)0, c, false, false);
    }

    const float bs = bias[ncol];
#pragma unroll
    for (int j = 0; j < 8; ++j) {
        const int m = m0 + j + ((lane >> 4) << 3);
        Y[m * Hd + ncol] = c[j] + bs;
    }
}

// ---------------------------------------------------------------------------
// Kernel 2: per (batch, 16-query tile):
//   scores[q,k] = sum_h wv[h]*tanh(Qh[q,h] + Kh[k,h])   (bv softmax-invariant)
//   attn = softmax_k(scores);  out = attn @ value[b]  (WMMA)
// ---------------------------------------------------------------------------
__global__ __launch_bounds__(256) void attn_kernel(
    const float* __restrict__ Qh,     // [B*LQ, 256]
    const float* __restrict__ Kh,     // [B*LK, 256]
    const float* __restrict__ wv,     // [256]
    const float* __restrict__ value,  // [B, LK, DV]
    float* __restrict__ out)          // [B, LQ, DV]
{
    const int t  = threadIdx.x;
    const int b  = blockIdx.x >> 5;
    const int q0 = (blockIdx.x & 31) * 16;

    extern __shared__ char smem[];
    _Float16* attn    = (_Float16*)(smem + ATTN_OFF);
    float*    red     = (float*)(smem + RED_OFF);
    float*    rowstat = (float*)(smem + ROWSTAT_OFF);
    float*    qr      = (float*)(smem + QR_OFF);
    float*    wvl     = (float*)(smem + WVL_OFF);
    float*    ktile   = (float*)(smem + KTILE_OFF);
    float*    sc      = (float*)(smem + SC_OFF);
    float*    vtile   = (float*)(smem + VTILE_OFF);

    // Stage 16 query-feature rows and wv (plain coalesced loads, done once)
    for (int idx = t; idx < 16 * 256; idx += 256)
        qr[idx] = Qh[(size_t)(b * LQ + q0 + (idx >> 8)) * Hd + (idx & 255)];
    wvl[t & 255] = wv[t & 255];
    __syncthreads();

    // ---- scores: 16 tiles of 32 keys, async-staged into LDS ---------------
    for (int kt = 0; kt < 16; ++kt) {
#pragma unroll
        for (int j = 0; j < 8; ++j) {                    // 32*256 f32 = 2048 x b128
            const int fidx = (t + j * 256) * 4;          // float idx in tile
            const int kk = fidx >> 8, hh = fidx & 255;
            async_b128(KTILE_OFF + (unsigned)(kk * KT_STRIDE + hh) * 4,
                       Kh + (size_t)(b * LK + kt * 32 + kk) * Hd + hh);
        }
        wait_async0();
        __syncthreads();

#pragma unroll
        for (int c2 = 0; c2 < 2; ++c2) {
            const int cell = t + c2 * 256;               // 512 cells = 16q x 32k
            const int q = cell >> 5, k = cell & 31;
            const v4f* qp = (const v4f*)(qr + q * 256);
            const v4f* kp = (const v4f*)(ktile + k * KT_STRIDE);
            const v4f* wp = (const v4f*)wvl;
            float s = 0.f;
#pragma unroll 4
            for (int h4 = 0; h4 < 64; ++h4) {            // ds_load_b128 x3 per 4 tanh
                const v4f qv = qp[h4], kv = kp[h4], wvv = wp[h4];
                s += wvv.x * fast_tanh(qv.x + kv.x);
                s += wvv.y * fast_tanh(qv.y + kv.y);
                s += wvv.z * fast_tanh(qv.z + kv.z);
                s += wvv.w * fast_tanh(qv.w + kv.w);
            }
            sc[q * 512 + kt * 32 + k] = s;
        }
        __syncthreads();
    }

    // ---- softmax over k per query row; thread = (q = t>>4, chunk j = t&15) -
    {
        const int q = t >> 4, j = t & 15;
        float mx = -INFINITY;
        for (int k = j * 32; k < j * 32 + 32; ++k) mx = fmaxf(mx, sc[q * 512 + k]);
        red[q * 16 + j] = mx;
        __syncthreads();
        if (j == 0) {
            float mm = red[q * 16];
            for (int i = 1; i < 16; ++i) mm = fmaxf(mm, red[q * 16 + i]);
            rowstat[q] = mm;
        }
        __syncthreads();
        const float rm = rowstat[q];
        float ssum = 0.f;
        for (int k = j * 32; k < j * 32 + 32; ++k) {
            const float e = __expf(sc[q * 512 + k] - rm);
            sc[q * 512 + k] = e;
            ssum += e;
        }
        red[q * 16 + j] = ssum;
        __syncthreads();
        if (j == 0) {
            float s = 0.f;
            for (int i = 0; i < 16; ++i) s += red[q * 16 + i];
            rowstat[q] = 1.f / s;
        }
        __syncthreads();
        const float inv = rowstat[q];
        for (int k = j * 32; k < j * 32 + 32; ++k)
            attn[q * 512 + k] = (_Float16)(sc[q * 512 + k] * inv);
        __syncthreads();   // attn done; qr/ktile/sc now dead -> vtile overlay ok
    }

    // ---- out[16,512] = attn[16,512] @ value[b][512,512] via WMMA -----------
    const int lane = t & 31;
    const int wave = t >> 5;                 // wave owns N-cols [wave*64, +64)
    const int lrow = lane & 15;
    const int kb   = (lane >> 4) * 8;
    const float* vb = value + (size_t)b * LK * DV;

    v8f acc[4] = {v8f{}, v8f{}, v8f{}, v8f{}};
    for (int k0 = 0; k0 < LK; k0 += 32) {
        // async-stage value[k0:k0+32, 0:512] f32 into vtile (64 KB, coalesced)
#pragma unroll
        for (int j = 0; j < 16; ++j) {                   // 32*512 f32 = 4096 x b128
            const int fidx = (t + j * 256) * 4;
            const int kk = fidx >> 9, dd = fidx & 511;
            async_b128(VTILE_OFF + (unsigned)fidx * 4,
                       vb + (size_t)(k0 + kk) * DV + dd);
        }
        wait_async0();
        __syncthreads();

        // A fragment: two 16B LDS reads from attn
        v16h a;
        {
            const v8h* ap8 = (const v8h*)(attn + lrow * 512 + k0 + kb);
            const v8h alo = ap8[0], ahi = ap8[2];        // +16 halfs
#pragma unroll
            for (int i = 0; i < 8; ++i) { a[i] = alo[i]; a[i + 8] = ahi[i]; }
        }
#pragma unroll
        for (int nt = 0; nt < 4; ++nt) {
            const int n = wave * 64 + nt * 16 + lrow;
            v16h bf;
#pragma unroll
            for (int i = 0; i < 8; ++i) {                // B[K][n] from vtile
                bf[i]     = (_Float16)vtile[(kb + i) * 512 + n];
                bf[i + 8] = (_Float16)vtile[(16 + kb + i) * 512 + n];
            }
            acc[nt] = __builtin_amdgcn_wmma_f32_16x16x32_f16(
                          false, a, false, bf, (short)0, acc[nt], false, false);
        }
        __syncthreads();   // before next chunk overwrites vtile
    }

    float* ob = out + (size_t)(b * LQ + q0) * DV;
#pragma unroll
    for (int nt = 0; nt < 4; ++nt) {
        const int n = wave * 64 + nt * 16 + lrow;
#pragma unroll
        for (int j = 0; j < 8; ++j) {
            const int m = j + ((lane >> 4) << 3);
            ob[m * DV + n] = acc[nt][j];
        }
    }
}

extern "C" void kernel_launch(void* const* d_in, const int* in_sizes, int n_in,
                              void* d_out, int out_size, void* d_ws, size_t ws_size,
                              hipStream_t stream) {
    const float* query = (const float*)d_in[0];
    const float* key   = (const float*)d_in[1];
    const float* value = (const float*)d_in[2];
    const float* wq    = (const float*)d_in[3];
    const float* bq    = (const float*)d_in[4];
    const float* wk    = (const float*)d_in[5];
    const float* bk    = (const float*)d_in[6];
    const float* wv    = (const float*)d_in[7];
    // d_in[8] = bv: softmax-invariant, unused.
    float* outp = (float*)d_out;

    // Workspace: Qh [2048,256] f32 then Kh [2048,256] f32 (4 MB total)
    float* Qh = (float*)d_ws;
    float* Kh = Qh + (size_t)Bsz * LQ * Hd;

    dim3 g1(Hd / 16, (Bsz * LQ) / 128);
    proj_kernel<<<g1, 256, 0, stream>>>(query, wq, bq, Qh);
    proj_kernel<<<g1, 256, 0, stream>>>(key,   wk, bk, Kh);

    attn_kernel<<<Bsz * (LQ / 16), 256, SMEM_BYTES, stream>>>(
        Qh, Kh, wv, value, outp);
}
